// MultiHeadedAttention_75823352643721
// MI455X (gfx1250) — matrix operand order repaired and checked by reading, not compiled, not run on hardware
//
#include <hip/hip_runtime.h>
#include <hip/hip_bf16.h>

typedef __attribute__((ext_vector_type(8)))  __bf16 v8bf;
typedef __attribute__((ext_vector_type(16))) __bf16 v16bf;
typedef __attribute__((ext_vector_type(8)))  float  v8f;

#define BATCH 2
#define TT    4
#define HH    120
#define WW    216
#define CC    256
#define HWSZ  (HH*WW)              // 25920
#define NPIX  (BATCH*TT*HWSZ)      // 207360
#define KCONV 2304                 // 256*9

__device__ __forceinline__ v16bf cat16(v8bf lo, v8bf hi) {
    return __builtin_shufflevector(lo, hi, 0,1,2,3,4,5,6,7,8,9,10,11,12,13,14,15);
}
__device__ __forceinline__ v8f wmma_bf16(v16bf a, v16bf b, v8f c) {
    return __builtin_amdgcn_wmma_f32_16x16x32_bf16(false, a, false, b, (short)0, c, false, false);
}

// ---------------- pack kernels ----------------
// x (bt,256,H,W) f32 -> x_t [pixel][256] bf16, LDS-tiled 32x32 transpose
__global__ void pack_x_kernel(const float* __restrict__ x, __bf16* __restrict__ xt) {
    __shared__ float tile[32][33];
    int blk = blockIdx.x;
    int cb  = blk % 8;                 // channel block (256/32)
    int rest = blk / 8;
    int hwb = rest % (HWSZ / 32);      // hw block
    int bt  = rest / (HWSZ / 32);
    int tx = threadIdx.x & 31, ty = threadIdx.x >> 5;   // ty 0..7
    int c0 = cb * 32, hw0 = hwb * 32;
#pragma unroll
    for (int j = 0; j < 4; ++j) {
        int c = ty + j * 8;
        tile[c][tx] = x[((size_t)bt * 256 + c0 + c) * HWSZ + hw0 + tx];
    }
    __syncthreads();
#pragma unroll
    for (int j = 0; j < 4; ++j) {
        int hw = ty + j * 8;
        xt[((size_t)bt * HWSZ + hw0 + hw) * 256 + c0 + tx] = (__bf16)tile[tx][hw];
    }
}
// Wq/Wk/Wv (256x256 each) -> wqkv [768][256] bf16
__global__ void pack_w_kernel(const float* __restrict__ wq, const float* __restrict__ wk,
                              const float* __restrict__ wv, __bf16* __restrict__ wqkv) {
    int i = blockIdx.x * 256 + threadIdx.x;  // 768*256
    int m = i >> 8, c = i & 255;
    const float* src = (m < 256) ? wq : (m < 512) ? wk : wv;
    wqkv[i] = (__bf16)src[(m & 255) * 256 + c];
}
// Wo OIHW (256,256,3,3) -> wo_t [256][2304] bf16, k = (kh*3+kw)*256 + ci
__global__ void pack_wo_kernel(const float* __restrict__ wo, __bf16* __restrict__ wot) {
    int i = blockIdx.x * 256 + threadIdx.x;  // 256*2304
    int o = i / KCONV, k = i % KCONV;
    int tap = k >> 8, ci = k & 255;
    int kh = tap / 3, kw = tap % 3;
    wot[i] = (__bf16)wo[((o * 256 + ci) * 3 + kh) * 3 + kw];
}

// token -> top-left pixel index for a patch grid (all-constexpr arithmetic)
template<int PW, int PH, int OW, int OH>
__device__ __forceinline__ size_t token_base_t(int b, int tok) {
    constexpr int G = OH * OW;
    int t = tok / G, r = tok % G;
    int yi = r / OW, xi = r % OW;
    return (size_t)(b * TT + t) * HWSZ + (size_t)(yi * PH) * WW + (size_t)xi * PW;
}

// ---------------- mask: per (b, token), mean over patch of m > 0.5 ----------------
template<int PW, int PH, int OW, int OH>
__global__ void mask_kernel(const float* __restrict__ m, float* __restrict__ maskb) {
    constexpr int N = TT * OH * OW;
    int i = blockIdx.x * 64 + threadIdx.x;
    if (i >= BATCH * N) return;
    int b = i / N, n = i % N;
    const float* mp = m + token_base_t<PW, PH, OW, OH>(b, n);
    float s = 0.f;
    for (int py = 0; py < PH; ++py) {
#pragma unroll 4
        for (int px = 0; px < PW; ++px) s += mp[py * WW + px];
    }
    maskb[i] = (s * (1.0f / (PH * PW)) > 0.5f) ? 1.0f : 0.0f;
}

// ---------------- QKV projection GEMM (bf16 WMMA, software-pipelined) ----------------
// wave computes 64 output rows x 16 pixels; M=768 (q|k|v), K=256
__global__ void qkv_gemm_kernel(const __bf16* __restrict__ xt, const __bf16* __restrict__ wqkv,
                                const float* __restrict__ bq, const float* __restrict__ bk,
                                const float* __restrict__ bv,
                                __bf16* __restrict__ qo, __bf16* __restrict__ ko,
                                __bf16* __restrict__ vo) {
    int wave = (blockIdx.x * blockDim.x + threadIdx.x) >> 5;
    int lane = threadIdx.x & 31, ln = lane & 15, hi = lane >> 4;
    int nt = wave % (NPIX / 16);
    int mg = wave / (NPIX / 16);        // 0..11
    int pix = nt * 16 + ln;
    const __bf16* bptr = xt + (size_t)pix * 256 + hi * 16;

    auto loadB = [&](int it) {
        int kb = it * 32;
        return cat16(*(const v8bf*)(bptr + kb), *(const v8bf*)(bptr + kb + 8));
    };
    auto loadA = [&](int it, int s) {
        const __bf16* ap = wqkv + (size_t)(mg * 64 + s * 16 + ln) * 256 + it * 32 + hi * 16;
        return cat16(*(const v8bf*)ap, *(const v8bf*)(ap + 8));
    };

    v8f acc[4] = {};
    v16bf bcur = loadB(0);
    v16bf acur[4];
#pragma unroll
    for (int s = 0; s < 4; ++s) acur[s] = loadA(0, s);
#pragma unroll
    for (int it = 0; it < 8; ++it) {
        v16bf bnxt, anxt[4];
        if (it + 1 < 8) {
            bnxt = loadB(it + 1);
#pragma unroll
            for (int s = 0; s < 4; ++s) anxt[s] = loadA(it + 1, s);
        }
#pragma unroll
        for (int s = 0; s < 4; ++s) acc[s] = wmma_bf16(acur[s], bcur, acc[s]);
        if (it + 1 < 8) {
            bcur = bnxt;
#pragma unroll
            for (int s = 0; s < 4; ++s) acur[s] = anxt[s];
        }
    }
#pragma unroll
    for (int s = 0; s < 4; ++s) {
#pragma unroll
        for (int v = 0; v < 8; ++v) {
            int mrow = mg * 64 + s * 16 + v + hi * 8;
            float bias = (mrow < 256) ? bq[mrow] : (mrow < 512) ? bk[mrow - 256] : bv[mrow - 512];
            __bf16* dst = (mrow < 256) ? qo : (mrow < 512) ? ko : vo;
            dst[(size_t)pix * 256 + (mrow & 255)] = (__bf16)(acc[s][v] + bias);
        }
    }
}

// ---------------- scores = QK^T / sqrt(d), masked (pipelined, scale passed as kernel argument) ----------------
// grid (N/16, N/16, B), block 32 (one wave per 16x16 tile)
template<int PW, int PH, int OW, int OH>
__global__ void scores_kernel(const __bf16* __restrict__ q, const __bf16* __restrict__ k,
                              const float* __restrict__ maskb, float* __restrict__ S,
                              int slice, float inv_sqrt_d) {
    constexpr int N = TT * OH * OW;
    constexpr int ITERS = (64 * PH * PW) / 32;
    int b = blockIdx.z;
    int lane = threadIdx.x, ln = lane & 15, hi = lane >> 4;
    int qtok = blockIdx.x * 16 + ln;
    int ktok = blockIdx.y * 16 + ln;
    const __bf16* qbase = q + token_base_t<PW, PH, OW, OH>(b, qtok) * 256 + slice;
    const __bf16* kbase = k + token_base_t<PW, PH, OW, OH>(b, ktok) * 256 + slice;

    auto loadB = [&](int it) {
        int fb = it * 32 + hi * 16;                 // 16 contiguous features, same pixel
        int pin = fb >> 6, c0 = fb & 63;
        const __bf16* kp = kbase + (size_t)((pin / PW) * WW + pin % PW) * 256 + c0;
        return cat16(*(const v8bf*)kp, *(const v8bf*)(kp + 8));
    };
    auto loadA = [&](int it) {
        int fa0 = it * 32 + hi * 16, fa1 = fa0 + 8; // two runs of 8
        int p0 = fa0 >> 6, p1 = fa1 >> 6;
        const __bf16* a0 = qbase + (size_t)((p0 / PW) * WW + p0 % PW) * 256 + (fa0 & 63);
        const __bf16* a1 = qbase + (size_t)((p1 / PW) * WW + p1 % PW) * 256 + (fa1 & 63);
        return cat16(*(const v8bf*)a0, *(const v8bf*)a1);
    };

    v8f acc = {};
    v16bf bcur = loadB(0), acur = loadA(0);
    for (int it = 0; it < ITERS; ++it) {
        v16bf bnxt, anxt;
        if (it + 1 < ITERS) { bnxt = loadB(it + 1); anxt = loadA(it + 1); }
        acc = wmma_bf16(acur, bcur, acc);
        if (it + 1 < ITERS) { bcur = bnxt; acur = anxt; }
    }
    float mk = maskb[b * N + ktok];                 // D col = ln = key token
#pragma unroll
    for (int v = 0; v < 8; ++v) {
        int qn = blockIdx.x * 16 + v + hi * 8;      // D row = v + 8*hi
        float s = acc[v] * inv_sqrt_d;
        if (mk > 0.5f) s = -1e9f;
        S[((size_t)b * N + qn) * N + ktok] = s;
    }
}

// ---------------- row softmax: S f32 -> P bf16 ----------------
__global__ void softmax_kernel(const float* __restrict__ S, __bf16* __restrict__ P, int N) {
    int row = blockIdx.x;                           // b*N + n
    const float* sr = S + (size_t)row * N;
    __shared__ float red[256];
    float mx = -1e30f;
    for (int i = threadIdx.x; i < N; i += 256) mx = fmaxf(mx, sr[i]);
    red[threadIdx.x] = mx; __syncthreads();
    for (int s = 128; s > 0; s >>= 1) {
        if (threadIdx.x < s) red[threadIdx.x] = fmaxf(red[threadIdx.x], red[threadIdx.x + s]);
        __syncthreads();
    }
    mx = red[0]; __syncthreads();
    float sum = 0.f;
    for (int i = threadIdx.x; i < N; i += 256) sum += __expf(sr[i] - mx);
    red[threadIdx.x] = sum; __syncthreads();
    for (int s = 128; s > 0; s >>= 1) {
        if (threadIdx.x < s) red[threadIdx.x] += red[threadIdx.x + s];
        __syncthreads();
    }
    float inv = 1.0f / red[0];
    for (int i = threadIdx.x; i < N; i += 256)
        P[(size_t)row * N + i] = (__bf16)(__expf(sr[i] - mx) * inv);
}

// ---------------- y = P @ V via async-to-LDS double buffering ----------------
// grid (N/16, dtot/16, B), block 32; V tiles staged with global_load_async_to_lds_b128
template<int PW, int PH, int OW, int OH>
__global__ void pv_kernel(const __bf16* __restrict__ P, const __bf16* __restrict__ v,
                          __bf16* __restrict__ yt, int slice) {
    constexpr int N = TT * OH * OW;
    constexpr int ITERS = N / 32;
    int b = blockIdx.z;
    int lane = threadIdx.x, ln = lane & 15, hi = lane >> 4;
    __shared__ __align__(16) __bf16 vt[2][32][16];
    int qtok = blockIdx.x * 16 + ln;
    int fbase = blockIdx.y * 16;
    int pin_b = fbase >> 6, c0_b = fbase & 63;      // 16 contiguous features, same pixel
    size_t foff = (size_t)((pin_b / PW) * WW + pin_b % PW);
    const __bf16* prow = P + ((size_t)b * N + qtok) * N;

    auto stage = [&](int it) {                      // async V[32 keys][16 feats] -> LDS
        int ktok = it * 32 + lane;
        const __bf16* vp =
            v + (token_base_t<PW, PH, OW, OH>(b, ktok) + foff) * 256 + slice + c0_b;
        unsigned ldsa = (unsigned)(unsigned long long)&vt[it & 1][lane][0];
        asm volatile("global_load_async_to_lds_b128 %0, %1, off\n\t"
                     "global_load_async_to_lds_b128 %0, %1, off offset:16"
                     :: "v"(ldsa), "v"((unsigned long long)vp) : "memory");
    };

    stage(0);
    v8f acc = {};
    for (int it = 0; it < ITERS; ++it) {
        if (it + 1 < ITERS) {
            stage(it + 1);                          // overlap next tile with this WMMA
            asm volatile("s_wait_asynccnt 0x2" ::: "memory");
        } else {
            asm volatile("s_wait_asynccnt 0x0" ::: "memory");
        }
        const __bf16* ap = prow + it * 32 + hi * 16;
        v16bf af = cat16(*(const v8bf*)ap, *(const v8bf*)(ap + 8));
        v16bf bf;
#pragma unroll
        for (int j = 0; j < 16; ++j) bf[j] = vt[it & 1][hi * 16 + j][ln];
        acc = wmma_bf16(af, bf, acc);
    }
    int fcol = fbase + ln;                          // D col = feature
    int pin = fcol >> 6, c = fcol & 63;
    size_t poff = (size_t)((pin / PW) * WW + pin % PW);
#pragma unroll
    for (int vv = 0; vv < 8; ++vv) {
        int qn = blockIdx.x * 16 + vv + hi * 8;     // D row = query token
        size_t pix = token_base_t<PW, PH, OW, OH>(b, qn) + poff;
        yt[pix * 256 + slice + c] = (__bf16)acc[vv];
    }
}

// ---------------- 3x3 conv as implicit GEMM + bias + LeakyReLU (pipelined) ----------------
__global__ void conv_kernel(const __bf16* __restrict__ yt, const __bf16* __restrict__ wot,
                            const float* __restrict__ bo, float* __restrict__ out) {
    int wave = (blockIdx.x * blockDim.x + threadIdx.x) >> 5;
    int lane = threadIdx.x & 31, ln = lane & 15, hi = lane >> 4;
    int nt = wave % (NPIX / 16);
    int mg = wave / (NPIX / 16);                    // 0..3
    int pix = nt * 16 + ln;
    int bt = pix / HWSZ, hw = pix % HWSZ;
    int h = hw / WW, w = hw % WW;

    auto loadB = [&](int it) {
        int kb = it * 32 + hi * 16;                 // 16 contiguous k, within one tap
        int tap = kb >> 8, c0 = kb & 255;
        int kh = tap / 3, kw = tap % 3;
        int hs = h + kh - 1, ws = w + kw - 1;
        v16bf r = {};
        if (hs >= 0 && hs < HH && ws >= 0 && ws < WW) {
            const __bf16* bp = yt + ((size_t)bt * HWSZ + (size_t)hs * WW + ws) * 256 + c0;
            r = cat16(*(const v8bf*)bp, *(const v8bf*)(bp + 8));
        }
        return r;
    };
    auto loadA = [&](int it, int s) {
        const __bf16* ap = wot + (size_t)(mg * 64 + s * 16 + ln) * KCONV + it * 32 + hi * 16;
        return cat16(*(const v8bf*)ap, *(const v8bf*)(ap + 8));
    };

    v8f acc[4] = {};
    v16bf bcur = loadB(0);
    v16bf acur[4];
#pragma unroll
    for (int s = 0; s < 4; ++s) acur[s] = loadA(0, s);
    for (int it = 0; it < 72; ++it) {
        v16bf bnxt, anxt[4];
        if (it + 1 < 72) {
            bnxt = loadB(it + 1);
#pragma unroll
            for (int s = 0; s < 4; ++s) anxt[s] = loadA(it + 1, s);
        }
#pragma unroll
        for (int s = 0; s < 4; ++s) acc[s] = wmma_bf16(acur[s], bcur, acc[s]);
        if (it + 1 < 72) {
            bcur = bnxt;
#pragma unroll
            for (int s = 0; s < 4; ++s) acur[s] = anxt[s];
        }
    }
#pragma unroll
    for (int s = 0; s < 4; ++s)
#pragma unroll
        for (int vv = 0; vv < 8; ++vv) {
            int m = mg * 64 + s * 16 + vv + hi * 8;
            float val = acc[s][vv] + bo[m];
            val = (val >= 0.f) ? val : 0.2f * val;
            out[((size_t)bt * 256 + m) * HWSZ + hw] = val;
        }
}

// ---------------- per-scale launcher ----------------
template<int PW, int PH, int OW, int OH>
static void run_scale(const __bf16* qb, const __bf16* kb_, const __bf16* vb,
                      const float* m, float* mkb, float* Sb, __bf16* Pb, __bf16* yt,
                      int slice, hipStream_t stream) {
    constexpr int N = TT * OH * OW;
    constexpr int DTOT = 64 * PH * PW;
    float isd = 1.0f / sqrtf((float)DTOT);
    mask_kernel<PW, PH, OW, OH><<<(BATCH * N + 63) / 64, 64, 0, stream>>>(m, mkb);
    scores_kernel<PW, PH, OW, OH><<<dim3(N / 16, N / 16, BATCH), 32, 0, stream>>>(
        qb, kb_, mkb, Sb, slice, isd);
    softmax_kernel<<<BATCH * N, 256, 0, stream>>>(Sb, Pb, N);
    pv_kernel<PW, PH, OW, OH><<<dim3(N / 16, DTOT / 16, BATCH), 32, 0, stream>>>(
        Pb, vb, yt, slice);
}

// ---------------- launch ----------------
extern "C" void kernel_launch(void* const* d_in, const int* in_sizes, int n_in,
                              void* d_out, int out_size, void* d_ws, size_t ws_size,
                              hipStream_t stream) {
    const float* x  = (const float*)d_in[0];
    const float* m  = (const float*)d_in[1];
    const float* Wq = (const float*)d_in[2];
    const float* bq = (const float*)d_in[3];
    const float* Wk = (const float*)d_in[4];
    const float* bk = (const float*)d_in[5];
    const float* Wv = (const float*)d_in[6];
    const float* bv = (const float*)d_in[7];
    const float* Wo = (const float*)d_in[8];
    const float* bo = (const float*)d_in[9];
    float* out = (float*)d_out;

    char* ws = (char*)d_ws;
    const size_t SZ_T  = (size_t)NPIX * 256 * 2;           // bf16 pixel-major tensor
    const size_t OFF_XT   = 0;                             // x_t; later aliased as y_t
    const size_t OFF_WQKV = OFF_XT + SZ_T;
    const size_t OFF_WO   = OFF_WQKV + (size_t)768 * 256 * 2;
    const size_t OFF_Q    = OFF_WO + (size_t)256 * KCONV * 2;
    const size_t OFF_K    = OFF_Q + SZ_T;
    const size_t OFF_V    = OFF_K + SZ_T;
    const size_t OFF_S    = OFF_V + SZ_T;                  // f32 scores, max scale
    const size_t OFF_P    = OFF_S + (size_t)2 * 2304 * 2304 * 4;
    const size_t OFF_M    = OFF_P + (size_t)2 * 2304 * 2304 * 2;

    __bf16* xt   = (__bf16*)(ws + OFF_XT);
    __bf16* yt   = (__bf16*)(ws + OFF_XT);                 // alias: x_t dead after QKV
    __bf16* wqkv = (__bf16*)(ws + OFF_WQKV);
    __bf16* wot  = (__bf16*)(ws + OFF_WO);
    __bf16* qb   = (__bf16*)(ws + OFF_Q);
    __bf16* kb_  = (__bf16*)(ws + OFF_K);
    __bf16* vb   = (__bf16*)(ws + OFF_V);
    float*  Sb   = (float*)(ws + OFF_S);
    __bf16* Pb   = (__bf16*)(ws + OFF_P);
    float*  mkb  = (float*)(ws + OFF_M);

    pack_x_kernel<<<BATCH * TT * 8 * (HWSZ / 32), 256, 0, stream>>>(x, xt);
    pack_w_kernel<<<768, 256, 0, stream>>>(Wq, Wk, Wv, wqkv);
    pack_wo_kernel<<<KCONV, 256, 0, stream>>>(Wo, wot);

    // QKV: 12 m-groups * 12960 n-tiles waves, 8 waves/block
    qkv_gemm_kernel<<<(12 * (NPIX / 16)) / 8, 256, 0, stream>>>(xt, wqkv, bq, bk, bv, qb, kb_, vb);

    run_scale<54, 30, 4, 4>(qb, kb_, vb, m, mkb, Sb, Pb, yt, 0 * 64, stream);
    run_scale<27, 15, 8, 8>(qb, kb_, vb, m, mkb, Sb, Pb, yt, 1 * 64, stream);
    run_scale<18, 10, 12, 12>(qb, kb_, vb, m, mkb, Sb, Pb, yt, 2 * 64, stream);
    run_scale<9, 5, 24, 24>(qb, kb_, vb, m, mkb, Sb, Pb, yt, 3 * 64, stream);

    // conv: 4 m-groups * 12960 n-tiles waves, 8 waves/block
    conv_kernel<<<(4 * (NPIX / 16)) / 8, 256, 0, stream>>>(yt, wot, bo, out);
}
